// EagerAttention_52948356825420
// MI455X (gfx1250) — matrix operand order repaired and checked by reading, not compiled, not run on hardware
//
#include <hip/hip_runtime.h>

typedef __bf16 bf16_t;
typedef __attribute__((ext_vector_type(16))) __bf16 bf16x16;
typedef __attribute__((ext_vector_type(8)))  float  f32x8;
typedef __attribute__((ext_vector_type(4)))  int    v4i;

#define WMMA_BF16(a, b, c) \
  __builtin_amdgcn_wmma_f32_16x16x32_bf16(false, (a), false, (b), (short)0, (c), false, false)

// ---- CDNA5 async global->LDS probes -------------------------------------
#if defined(__has_builtin)
#if __has_builtin(__builtin_amdgcn_global_load_async_to_lds_b128)
#define HAVE_ASYNC_LDS 1
#endif
#if __has_builtin(__builtin_amdgcn_s_wait_asynccnt)
#define HAVE_WAIT_ASYNC 1
#endif
#endif

#define AS1 __attribute__((address_space(1)))
#define AS3 __attribute__((address_space(3)))

// 16-byte global->LDS copy: async (no VGPR round trip, ASYNCcnt tracked) when
// the toolchain has the gfx1250 builtin, else synchronous fallback.
// Builtin signature (from probe diagnostic): param0 is a non-const pointer to
// v4i (int __vector_size__(16)); pass AS1/AS3-qualified pointers (implicitly
// convertible to generic if the params are generic).
__device__ __forceinline__ void cp16_g2l(const bf16_t* g, bf16_t* l) {
#ifdef HAVE_ASYNC_LDS
  v4i* gp = (v4i*)g;  // drop const, retype (generic)
  v4i* lp = (v4i*)l;
  __builtin_amdgcn_global_load_async_to_lds_b128((AS1 v4i*)gp, (AS3 v4i*)lp, 0, 0);
#else
  *reinterpret_cast<float4*>(l) = *reinterpret_cast<const float4*>(g);
#endif
}
__device__ __forceinline__ void wait_async0() {
#ifdef HAVE_ASYNC_LDS
#ifdef HAVE_WAIT_ASYNC
  __builtin_amdgcn_s_wait_asynccnt(0);
#else
  asm volatile("s_wait_asynccnt 0x0" ::: "memory");
#endif
#endif
}

// ---------------------------------------------------------------------------
// Fragment loaders. Per CDNA5 ISA (CDNA5 ISA):
//  A operand (16x32 bf16): lane half h holds K = {8h..8h+7} U {16+8h..16+8h+7}
//  B operand (32x16 bf16): lane half h holds K = {8h..8h+7} U {16+8h..16+8h+7} for column lane%16
// ---------------------------------------------------------------------------
__device__ __forceinline__ bf16x16 frag_ld(const bf16_t* p0, const bf16_t* p1) {
  union { bf16x16 v; float4 f[2]; } u;
  u.f[0] = *reinterpret_cast<const float4*>(p0);
  u.f[1] = *reinterpret_cast<const float4*>(p1);
  return u.v;
}
__device__ __forceinline__ bf16x16 frag_a(const bf16_t* row, int half) {
  return frag_ld(row + half * 8, row + 16 + half * 8);
}
__device__ __forceinline__ bf16x16 frag_b(const bf16_t* row, int half) {
  return frag_ld(row + half * 8, row + 16 + half * 8);
}

// ---------------------------------------------------------------------------
// Elementwise fp32 -> bf16
// ---------------------------------------------------------------------------
__global__ __launch_bounds__(256) void f32_to_bf16_kernel(
    const float* __restrict__ in, bf16_t* __restrict__ out, int n) {
  for (int i = blockIdx.x * 256 + threadIdx.x; i < n; i += gridDim.x * 256)
    out[i] = (bf16_t)in[i];
}

// ---------------------------------------------------------------------------
// C[M,N] (fp32) = A[M,K] (bf16, row-major) * B[N,K]^T (bf16, row-major)
// Block tile 128x128, BK=32, 256 threads = 8 waves, each wave 32x64
// (2 A-frags x 4 B-frags = 8 WMMA per k-step). Double-buffered LDS with
// async global->LDS prefetch of tile i+1 overlapped with WMMAs of tile i.
// ---------------------------------------------------------------------------
#define BM 128
#define BN 128
#define BK 32

__global__ __launch_bounds__(256) void gemm_bf16_nt(
    const bf16_t* __restrict__ A, const bf16_t* __restrict__ B,
    float* __restrict__ C, int M, int N, int K) {
  __shared__ __align__(16) bf16_t As[2][BM][BK + 8];
  __shared__ __align__(16) bf16_t Bs[2][BN][BK + 8];
  const int tid  = threadIdx.x;
  const int lane = tid & 31;
  const int w    = tid >> 5;
  const int half = lane >> 4;
  const int l16  = lane & 15;
  const int wm   = w & 3;   // 0..3 -> 32-row slabs
  const int wn   = w >> 2;  // 0..1 -> 64-col slabs
  const int bm = blockIdx.y * BM, bn = blockIdx.x * BN;

  // Global load indices (32-bit; all buffers < 2^31 elements)
  const int arow = tid >> 2, akoff = (tid & 3) * 8;

  // Issue one block-tile fill (4x 16B per thread) into LDS buffer `buf`.
  auto issue = [&](int buf, int k0) {
#pragma unroll
    for (int i = 0; i < 2; ++i) {
      const int row = arow + i * 64;
      cp16_g2l(&A[(bm + row) * K + k0 + akoff], &As[buf][row][akoff]);
      cp16_g2l(&B[(bn + row) * K + k0 + akoff], &Bs[buf][row][akoff]);
    }
  };

  f32x8 c[2][4] = {};
  const int nk = K / BK;
  issue(0, 0);

  for (int i = 0; i < nk; ++i) {
    const int cur = i & 1;
    wait_async0();     // this wave's fills of buf `cur` are in LDS
    __syncthreads();   // ... and everyone else's
    if (i + 1 < nk) issue(cur ^ 1, (i + 1) * BK);  // prefetch next tile

    bf16x16 af[2], bfv[4];
#pragma unroll
    for (int s = 0; s < 2; ++s)
      af[s] = frag_a(&As[cur][wm * 32 + s * 16 + l16][0], half);
#pragma unroll
    for (int s = 0; s < 4; ++s)
      bfv[s] = frag_b(&Bs[cur][wn * 64 + s * 16 + l16][0], half);
#pragma unroll
    for (int sm = 0; sm < 2; ++sm)
#pragma unroll
      for (int sn = 0; sn < 4; ++sn)
        c[sm][sn] = WMMA_BF16(af[sm], bfv[sn], c[sm][sn]);
    __syncthreads();   // buf `cur` free before it is refilled at i+2
  }

  // C/D layout: VGPR r holds M = r + 8*(lane/16), N = lane%16.
#pragma unroll
  for (int sm = 0; sm < 2; ++sm) {
#pragma unroll
    for (int sn = 0; sn < 4; ++sn) {
      const int col  = bn + wn * 64 + sn * 16 + l16;
      const int row0 = bm + wm * 32 + sm * 16 + 8 * half;
      const int base = row0 * N + col;
#pragma unroll
      for (int r = 0; r < 8; ++r)
        C[base + r * N] = c[sm][sn][r];
    }
  }
}

// ---------------------------------------------------------------------------
// RoPE + relayout: in [b][s][nheads*64] fp32 -> out [b][head][s][64] bf16
// One wave per (b,head,s) row; lane d in 0..31 handles the (d, d+32) pair.
// ---------------------------------------------------------------------------
__global__ __launch_bounds__(256) void rope_kernel(
    const float* __restrict__ in, bf16_t* __restrict__ out, int nheads) {
  const int w = threadIdx.x >> 5;
  const int d = threadIdx.x & 31;
  const long gw = (long)blockIdx.x * 8 + w;
  const int s = (int)(gw % 2048);
  const int h = (int)((gw / 2048) % nheads);
  const int b = (int)(gw / ((long)2048 * nheads));
  const float* src = in + ((b * 2048 + s) * nheads + h) * 64;
  const float x0 = src[d], x1 = src[d + 32];
  const float invf = __powf(10000.0f, -(float)d * (1.0f / 32.0f));
  const float ang = (float)s * invf;
  float sn, cs;
  sincosf(ang, &sn, &cs);
  bf16_t* dst = out + ((b * nheads + h) * 2048 + s) * 64;
  dst[d]      = (bf16_t)(x0 * cs - x1 * sn);
  dst[d + 32] = (bf16_t)(x1 * cs + x0 * sn);
}

// ---------------------------------------------------------------------------
// V relayout+transpose: vf [b][s][8*64] fp32 -> vt [b][kvh][d=64][s=2048] bf16
// ---------------------------------------------------------------------------
__global__ __launch_bounds__(256) void vtrans_kernel(
    const float* __restrict__ vf, bf16_t* __restrict__ vt) {
  const int idx = blockIdx.x * 256 + threadIdx.x;  // total 2*8*64*2048 = 2^21
  const int s   = idx & 2047;
  const int d   = (idx >> 11) & 63;
  const int kvh = (idx >> 17) & 7;
  const int b   = idx >> 20;
  vt[idx] = (bf16_t)vf[(b * 2048 + s) * 512 + kvh * 64 + d];
}

// ---------------------------------------------------------------------------
// Flash attention, causal. One wave per 16-query tile, 32-key blocks.
// qb [b][32][2048][64], kb [b][8][2048][64], vt [b][8][64][2048] (all bf16)
// attnb [b][s][head*64+d] bf16
// ---------------------------------------------------------------------------
__global__ __launch_bounds__(256) void attn_kernel(
    const bf16_t* __restrict__ qb, const bf16_t* __restrict__ kb,
    const bf16_t* __restrict__ vt, bf16_t* __restrict__ attnb) {
  __shared__ __align__(16) bf16_t Ps[8][16][40];  // per-wave P staging
  const int tid  = threadIdx.x;
  const int w    = tid >> 5;
  const int lane = tid & 31;
  const int half = lane >> 4;
  const int l16  = lane & 15;
  const int h = blockIdx.y, b = blockIdx.z;
  const int kvh = h >> 2;  // 32 heads -> 8 kv heads (groups of 4)
  const int q0 = (blockIdx.x * 8 + w) * 16;

  const bf16_t* qh = qb + (b * 32 + h) * (2048 * 64);
  const bf16_t* kh = kb + (b * 8 + kvh) * (2048 * 64);
  const bf16_t* vh = vt + (b * 8 + kvh) * (64 * 2048);

  const bf16_t* qrow = qh + (q0 + l16) * 64;
  const bf16x16 qf0 = frag_a(qrow, half);       // d 0..31
  const bf16x16 qf1 = frag_a(qrow + 32, half);  // d 32..63

  float mrow[8], lrow[8];
#pragma unroll
  for (int r = 0; r < 8; ++r) { mrow[r] = -3.0e38f; lrow[r] = 0.0f; }
  f32x8 o[4] = {};

  for (int k0 = 0; k0 < q0 + 16; k0 += 32) {
    // ---- scores: two 16x16 C tiles over 32 keys, K-dim = 64 (two d-halves)
    f32x8 s0 = {}, s1 = {};
    {
      const bf16_t* kr0 = kh + (k0 + l16) * 64;
      s0 = WMMA_BF16(qf0, frag_b(kr0, half), s0);
      s0 = WMMA_BF16(qf1, frag_b(kr0 + 32, half), s0);
      const bf16_t* kr1 = kr0 + 16 * 64;
      s1 = WMMA_BF16(qf0, frag_b(kr1, half), s1);
      s1 = WMMA_BF16(qf1, frag_b(kr1 + 32, half), s1);
    }
    // ---- mask + online softmax (row reductions across half-wave, wave32)
    const int kc0 = k0 + l16, kc1 = k0 + 16 + l16;
#pragma unroll
    for (int r = 0; r < 8; ++r) {
      const int qi = q0 + r + 8 * half;
      const float v0 = s0[r] * 0.125f + (kc0 > qi ? -1.0e9f : 0.0f);
      const float v1 = s1[r] * 0.125f + (kc1 > qi ? -1.0e9f : 0.0f);
      float mx = fmaxf(v0, v1);
#pragma unroll
      for (int off = 1; off < 16; off <<= 1)
        mx = fmaxf(mx, __shfl_xor(mx, off, 32));
      const float mnew = fmaxf(mrow[r], mx);
      const float corr = __expf(mrow[r] - mnew);
      mrow[r] = mnew;
      const float p0 = __expf(v0 - mnew);
      const float p1 = __expf(v1 - mnew);
      float rs = p0 + p1;
#pragma unroll
      for (int off = 1; off < 16; off <<= 1)
        rs += __shfl_xor(rs, off, 32);
      lrow[r] = lrow[r] * corr + rs;
#pragma unroll
      for (int t = 0; t < 4; ++t) o[t][r] *= corr;
      // C-layout -> LDS (row M = r + 8*half, cols lane%16 and 16+lane%16)
      Ps[w][r + 8 * half][l16]      = (bf16_t)p0;
      Ps[w][r + 8 * half][16 + l16] = (bf16_t)p1;
    }
    asm volatile("s_wait_dscnt 0x0" ::: "memory");  // LDS stores visible to reload
    // ---- P (A operand) x V (B operand; vt rows [d][s] contiguous in k)
    const bf16x16 pf = frag_a(&Ps[w][l16][0], half);
#pragma unroll
    for (int t = 0; t < 4; ++t) {
      const bf16_t* vr = vh + (t * 16 + l16) * 2048 + k0;
      o[t] = WMMA_BF16(pf, frag_b(vr, half), o[t]);
    }
  }

  // ---- normalize, write bf16 in [b][s][head*64+d]; 32-bit indexing with
  // constant +2048 per row so stores use immediate offsets.
#pragma unroll
  for (int t = 0; t < 4; ++t) {
    const int col  = h * 64 + t * 16 + l16;
    const int base = (b * 2048 + q0 + 8 * half) * 2048 + col;
#pragma unroll
    for (int r = 0; r < 8; ++r)
      attnb[base + r * 2048] = (bf16_t)(o[t][r] / lrow[r]);
  }
}

// ---------------------------------------------------------------------------
// Host-side orchestration (all on `stream`, graph-capture safe)
// Inputs: 0=x(f32), 1=position_ids(arange, unused), 2=mask(causal, applied
// analytically), 3=wq, 4=wk, 5=wv, 6=wo
// ---------------------------------------------------------------------------
extern "C" void kernel_launch(void* const* d_in, const int* in_sizes, int n_in,
                              void* d_out, int out_size, void* d_ws, size_t ws_size,
                              hipStream_t stream) {
  (void)in_sizes; (void)n_in; (void)out_size; (void)ws_size;
  const float* x  = (const float*)d_in[0];
  const float* wq = (const float*)d_in[3];
  const float* wk = (const float*)d_in[4];
  const float* wv = (const float*)d_in[5];
  const float* wo = (const float*)d_in[6];
  float* out = (float*)d_out;

  char* ws = (char*)d_ws;
  size_t off = 0;
  auto carve = [&](size_t bytes) { char* p = ws + off; off += bytes; return p; };
  bf16_t* xb   = (bf16_t*)carve(16777216);  // 4096x2048 bf16
  bf16_t* wqb  = (bf16_t*)carve(8388608);   // 2048x2048 bf16
  bf16_t* wkb  = (bf16_t*)carve(2097152);   // 512x2048 bf16
  bf16_t* wvb  = (bf16_t*)carve(2097152);
  bf16_t* wob  = (bf16_t*)carve(8388608);
  float*  qf   = (float*)carve(33554432);   // 4096x2048 f32 (reused as attnb)
  float*  kf   = (float*)carve(8388608);    // 4096x512 f32
  float*  vf   = (float*)carve(8388608);
  bf16_t* qbuf = (bf16_t*)carve(16777216);  // [2][32][2048][64] bf16
  bf16_t* kbuf = (bf16_t*)carve(4194304);   // [2][8][2048][64] bf16
  bf16_t* vtb  = (bf16_t*)carve(4194304);   // [2][8][64][2048] bf16
  bf16_t* attnb = (bf16_t*)qf;              // qf dead after RoPE; alias

  // Stage 0: bf16 conversions
  f32_to_bf16_kernel<<<4096, 256, 0, stream>>>(x,  xb,  8388608);
  f32_to_bf16_kernel<<<4096, 256, 0, stream>>>(wq, wqb, 4194304);
  f32_to_bf16_kernel<<<2048, 256, 0, stream>>>(wk, wkb, 1048576);
  f32_to_bf16_kernel<<<2048, 256, 0, stream>>>(wv, wvb, 1048576);
  f32_to_bf16_kernel<<<4096, 256, 0, stream>>>(wo, wob, 4194304);

  // Stage 1: QKV projections (WMMA)
  gemm_bf16_nt<<<dim3(2048 / BN, 4096 / BM), 256, 0, stream>>>(xb, wqb, qf, 4096, 2048, 2048);
  gemm_bf16_nt<<<dim3(512 / BN, 4096 / BM), 256, 0, stream>>>(xb, wkb, kf, 4096, 512, 2048);
  gemm_bf16_nt<<<dim3(512 / BN, 4096 / BM), 256, 0, stream>>>(xb, wvb, vf, 4096, 512, 2048);

  // Stage 2: RoPE + head-major relayout, V transpose
  rope_kernel<<<(2 * 32 * 2048) / 8, 256, 0, stream>>>(qf, qbuf, 32);
  rope_kernel<<<(2 * 8 * 2048) / 8, 256, 0, stream>>>(kf, kbuf, 8);
  vtrans_kernel<<<8192, 256, 0, stream>>>(vf, vtb);

  // Stage 3: causal flash attention (WMMA)
  attn_kernel<<<dim3(16, 32, 2), 256, 0, stream>>>(qbuf, kbuf, vtb, attnb);

  // Stage 4: output projection (WMMA) -> fp32 d_out
  gemm_bf16_nt<<<dim3(2048 / BN, 4096 / BM), 256, 0, stream>>>(attnb, wob, out, 4096, 2048, 2048);
}
